// ReadingLayer_13563506721325
// MI455X (gfx1250) — matrix operand order repaired and checked by reading, not compiled, not run on hardware
//
#include <hip/hip_runtime.h>
#include <hip/hip_bf16.h>
#include <math.h>

#define BB   16
#define SS   2048
#define II   256
#define HH   512
#define DD   32          // feedback delay == time-block size
#define KCAT 768         // II + HH
#define KT1  24          // KCAT/32  (K-tiles, phase-2 GEMM)
#define KT2  16          // HH/32    (K-tiles, phase-4 GEMM)
#define NT   32          // HH/16    (N-tiles)

typedef __attribute__((ext_vector_type(16))) __bf16   bf16x16;
typedef __attribute__((ext_vector_type(8)))  float    f32x8;
typedef __attribute__((ext_vector_type(4)))  unsigned u32x4;

union Frag { bf16x16 v; u32x4 q[2]; };

// ---------------- one-time layout/precision conversion ----------------

// x (f32) -> x (bf16), row-major [B,S,I]
__global__ void cvt_x_kernel(const float* __restrict__ x,
                             __hip_bfloat16* __restrict__ xbf) {
  size_t i = ((size_t)blockIdx.x * blockDim.x + threadIdx.x) * 4;
  float4 f = *(const float4*)(x + i);
  xbf[i + 0] = __float2bfloat16(f.x);
  xbf[i + 1] = __float2bfloat16(f.y);
  xbf[i + 2] = __float2bfloat16(f.z);
  xbf[i + 3] = __float2bfloat16(f.w);
}

// W [H, KCAT] -> W^T swizzled as WMMA B-operand tiles: [nt][kt][lane][16]
// B element (k, n) = W[n*KCAT + k]; lane holds column n = nt*16 + (lane&15),
// k = kt*32 + ((lane&16)?8:0) + (j&7) + ((j&8)?16:0).
__global__ void swz_W_kernel(const float* __restrict__ W,
                             __hip_bfloat16* __restrict__ WTsw) {
  int tid  = blockIdx.x * blockDim.x + threadIdx.x;   // 32*24*32*16 = 393216
  int j    = tid & 15;
  int lane = (tid >> 4) & 31;
  int rest = tid >> 9;
  int kt   = rest % KT1;
  int nt   = rest / KT1;
  int k = kt * 32 + ((lane & 16) ? 8 : 0) + (j & 7) + ((j & 8) ? 16 : 0);
  int n = nt * 16 + (lane & 15);
  WTsw[tid] = __float2bfloat16(W[(size_t)n * KCAT + k]);
}

// mem [B,H,H] -> per-batch B operand for ikv GEMM: B(k=h, n=v) = mem[b][v][h]
// layout [b][nt][kt][lane][16]
__global__ void swz_mem_kernel(const float* __restrict__ mem,
                               __hip_bfloat16* __restrict__ memTsw) {
  int tid  = blockIdx.x * blockDim.x + threadIdx.x;   // 16*32*16*32*16 = 4194304
  int j    = tid & 15;
  int lane = (tid >> 4) & 31;
  int kt   = (tid >> 9) & 15;
  int nt   = (tid >> 13) & 31;
  int b    = tid >> 18;
  int h = kt * 32 + ((lane & 16) ? 8 : 0) + (j & 7) + ((j & 8) ? 16 : 0);
  int v = nt * 16 + (lane & 15);
  memTsw[tid] = __float2bfloat16(mem[((size_t)b * HH + v) * HH + h]);
}

// ---------------- persistent recurrent kernel: 1 workgroup per batch ----------------

__global__ __launch_bounds__(1024) void recurrent_kernel(
    const __hip_bfloat16* __restrict__ xbf,     // [B,S,I] bf16
    const __hip_bfloat16* __restrict__ WTsw,    // swizzled W^T  (K=768, N=512)
    const __hip_bfloat16* __restrict__ memTsw,  // swizzled mem^T per batch (K=512, N=512)
    float* __restrict__ keys,                   // [B,S,H]
    float* __restrict__ vals)                   // [B,S,H]
{
  // LDS: 49,664 + 66,048 + 33,280 + 4,096 = 153,088 bytes
  __shared__ __hip_bfloat16 cat[DD][776];     // [t][0:256)=x_t, [256:768)=fb (prev block vals)
  __shared__ float          iblk[DD][516];    // GEMM results (i, then reused for ikv)
  __shared__ __hip_bfloat16 keyblk[DD][520];  // key_out block (A operand of ikv GEMM)
  __shared__ float          kvs[HH];
  __shared__ float          vvs[HH];

  const int b    = blockIdx.x;
  const int tid  = threadIdx.x;
  const int lane = tid & 31;
  const int wave = tid >> 5;                  // 0..31 == N-tile owned by this wave

  // init states and feedback region (circular buffer starts at zero)
  for (int i = tid; i < HH; i += 1024) { kvs[i] = 0.0f; vvs[i] = 0.0f; }
  for (int i = tid; i < DD * HH; i += 1024)
    cat[i >> 9][II + (i & 511)] = __float2bfloat16(0.0f);
  __syncthreads();

  const __hip_bfloat16* memB = memTsw + (size_t)b * (NT * KT2 * 32 * 16);

  const int row0 = (lane & 15);               // A row, M-tile 0
  const int row1 = 16 + (lane & 15);          // A row, M-tile 1
  const int ksel = (lane & 16) ? 8 : 0;       // A-fragment K-half select
  const int rb   = (lane & 16) ? 8 : 0;       // C/D row base within M-tile
  const int col  = wave * 16 + (lane & 15);   // C/D column

  #pragma unroll 1
  for (int jb = 0; jb < SS / DD; ++jb) {
    const int t0 = jb * DD;

    // ---- phase 1: stage x_t rows into cat[:, 0:256) (8192 bf16, 8 per thread)
    {
      int idx = tid * 8;
      int t = idx >> 8;
      int k = idx & 255;
      u32x4 src = *(const u32x4*)(xbf + ((size_t)(b * SS + t0 + t) * II + k));
      *(u32x4*)(&cat[t][k]) = src;
    }
    __syncthreads();

    // ---- phase 2: i = [x, fb] @ W^T   (M=32, N=512, K=768) -> iblk
    // wave owns nt = wave; both M-tiles share each B fragment.
    {
      f32x8 acc0 = {}, acc1 = {};
      #pragma unroll 1
      for (int kt = 0; kt < KT1; ++kt) {
        Frag a0, a1, bf;
        const int kb = kt * 32 + ksel;
        const __hip_bfloat16* bp = WTsw + ((size_t)(wave * KT1 + kt) * 32 + lane) * 16;
        bf.q[0] = *(const u32x4*)(bp);
        bf.q[1] = *(const u32x4*)(bp + 8);
        a0.q[0] = *(const u32x4*)(&cat[row0][kb]);
        a0.q[1] = *(const u32x4*)(&cat[row0][kb + 16]);
        a1.q[0] = *(const u32x4*)(&cat[row1][kb]);
        a1.q[1] = *(const u32x4*)(&cat[row1][kb + 16]);
        acc0 = __builtin_amdgcn_wmma_f32_16x16x32_bf16(false, a0.v, false, bf.v,
                                                       (short)0, acc0, false, false);
        acc1 = __builtin_amdgcn_wmma_f32_16x16x32_bf16(false, a1.v, false, bf.v,
                                                       (short)0, acc1, false, false);
      }
      #pragma unroll
      for (int r = 0; r < 8; ++r) {
        iblk[rb + r][col]      = acc0[r];
        iblk[16 + rb + r][col] = acc1[r];
      }
    }
    __syncthreads();

    // ---- phase 3: kv leaky-integrator scan + tanh -> keys, keyblk (bf16)
    if (tid < HH) {
      const int h = tid;
      float kvv = kvs[h];
      #pragma unroll 4
      for (int t = 0; t < DD; ++t) {
        kvv = 0.9f * kvv + 0.1f * iblk[t][h];
        float ko = tanhf(kvv);
        keys[((size_t)b * SS + t0 + t) * HH + h] = ko;
        keyblk[t][h] = __float2bfloat16(ko);
      }
      kvs[h] = kvv;
    } else if (jb + 1 < SS / DD) {
      // idle upper half: prefetch next block's x rows into cache (global_prefetch_b8)
      int p = tid - 512;                       // 0..511
      int t = p >> 4;                          // 0..31
      int k = (p & 15) * 16;                   // 0..240
      __builtin_prefetch(xbf + ((size_t)(b * SS + t0 + DD + t) * II + k), 0, 0);
    }
    __syncthreads();

    // ---- phase 4: ikv = key @ mem[b]^T  (M=32, N=512, K=512) -> iblk (reuse)
    {
      f32x8 acc0 = {}, acc1 = {};
      #pragma unroll 1
      for (int kt = 0; kt < KT2; ++kt) {
        Frag a0, a1, bf;
        const int kb = kt * 32 + ksel;
        const __hip_bfloat16* bp = memB + ((size_t)(wave * KT2 + kt) * 32 + lane) * 16;
        bf.q[0] = *(const u32x4*)(bp);
        bf.q[1] = *(const u32x4*)(bp + 8);
        a0.q[0] = *(const u32x4*)(&keyblk[row0][kb]);
        a0.q[1] = *(const u32x4*)(&keyblk[row0][kb + 16]);
        a1.q[0] = *(const u32x4*)(&keyblk[row1][kb]);
        a1.q[1] = *(const u32x4*)(&keyblk[row1][kb + 16]);
        acc0 = __builtin_amdgcn_wmma_f32_16x16x32_bf16(false, a0.v, false, bf.v,
                                                       (short)0, acc0, false, false);
        acc1 = __builtin_amdgcn_wmma_f32_16x16x32_bf16(false, a1.v, false, bf.v,
                                                       (short)0, acc1, false, false);
      }
      #pragma unroll
      for (int r = 0; r < 8; ++r) {
        iblk[rb + r][col]      = acc0[r];
        iblk[16 + rb + r][col] = acc1[r];
      }
    }
    __syncthreads();

    // ---- phase 5: vv scan + tanh -> vals, feedback for next block
    if (tid < HH) {
      const int h = tid;
      float vvv = vvs[h];
      #pragma unroll 4
      for (int t = 0; t < DD; ++t) {
        vvv = 0.9f * vvv + 0.1f * iblk[t][h];
        float vo = tanhf(vvv);
        vals[((size_t)b * SS + t0 + t) * HH + h] = vo;
        cat[t][II + h] = __float2bfloat16(vo);   // becomes fb of block jb+1
      }
      vvs[h] = vvv;
    }
    __syncthreads();
  }
}

// ---------------- launch ----------------

extern "C" void kernel_launch(void* const* d_in, const int* in_sizes, int n_in,
                              void* d_out, int out_size, void* d_ws, size_t ws_size,
                              hipStream_t stream) {
  const float* x   = (const float*)d_in[0];   // [16,2048,256]
  const float* mem = (const float*)d_in[1];   // [16,512,512]
  const float* W   = (const float*)d_in[2];   // [512,768]

  float* keys = (float*)d_out;
  float* vals = keys + (size_t)BB * SS * HH;

  char* ws = (char*)d_ws;
  __hip_bfloat16* xbf    = (__hip_bfloat16*)(ws);                         // 16,777,216 B
  __hip_bfloat16* WTsw   = (__hip_bfloat16*)(ws + 16777216);              //    786,432 B
  __hip_bfloat16* memTsw = (__hip_bfloat16*)(ws + 16777216 + 786432);     //  8,388,608 B

  // conversions (re-run every call: deterministic, no caching)
  cvt_x_kernel<<<(BB * SS * II) / (4 * 256), 256, 0, stream>>>(x, xbf);
  swz_W_kernel<<<(NT * KT1 * 32 * 16) / 256, 256, 0, stream>>>(W, WTsw);
  swz_mem_kernel<<<(BB * NT * KT2 * 32 * 16) / 256, 256, 0, stream>>>(mem, memTsw);

  // one workgroup per batch; all recurrence handled with intra-WG barriers
  recurrent_kernel<<<BB, 1024, 0, stream>>>(xbf, WTsw, memTsw, keys, vals);
}
